// MultiHeadAttention_33629593928181
// MI455X (gfx1250) — compile-verified, matrix operand order checked by reading, not run on hardware
//
#include <hip/hip_runtime.h>
#include <stdint.h>

typedef __attribute__((ext_vector_type(16))) __bf16 v16bf;
typedef __attribute__((ext_vector_type(8)))  float  v8f;
typedef int v4i_vs __attribute__((vector_size(16)));   // matches builtin's V4i

#define B_  4
#define T_  2048
#define C_  1024
#define H_  16
#define D_  64

#define BM 128
#define BN 128
#define BK 32
#define LPAD 40   // padded k-stride (ushorts) for LDS tiles

#if __has_builtin(__builtin_amdgcn_global_load_async_to_lds_b128)
#define HAVE_ASYNC_LDS 1
#else
#define HAVE_ASYNC_LDS 0
#endif

#if __has_builtin(__builtin_amdgcn_s_wait_asynccnt)
#define WAIT_ASYNC(n) __builtin_amdgcn_s_wait_asynccnt(n)
#else
#define WAIT_ASYNC(n) asm volatile("s_wait_asynccnt %0" :: "i"(n) : "memory")
#endif

__device__ inline unsigned short f2bf(float f) {
  union { float f; uint32_t u; } x; x.f = f;
  uint32_t u = x.u;
  uint32_t r = (u + 0x7fffu + ((u >> 16) & 1u)) >> 16;  // RNE
  return (unsigned short)r;
}

// 16-byte global -> LDS copy. Async (ASYNCcnt) on CDNA5, register fallback otherwise.
// AS1 ptr == 64-bit global addr; AS3 ptr == low 32 bits of generic shared addr
// (ISA: LDS aperture truncates to addr[31:0]), so int->ptr casts are exact.
__device__ inline void cp16_g2l(const unsigned short* g, unsigned short* l) {
#if HAVE_ASYNC_LDS
  __builtin_amdgcn_global_load_async_to_lds_b128(
      (v4i_vs __attribute__((address_space(1)))*)(uintptr_t)g,
      (v4i_vs __attribute__((address_space(3)))*)(uint32_t)(uintptr_t)l, 0, 0);
#else
  *(uint4*)l = *(const uint4*)g;
#endif
}

// Load one 16x32 bf16 WMMA operand fragment from padded LDS tile [row][k]:
//   lane<16 : row = base+lane,    K = 0..7 (v0-3), 16..23 (v4-7)
//   lane>=16: row = base+lane-16, K = 8..15,       24..31
__device__ inline v16bf frag_ld(const unsigned short* base, int row, int stride, int lane) {
  int r    = row + (lane & 15);
  int koff = (lane >> 4) * 8;
  union { uint4 u[2]; v16bf v; } f;
  f.u[0] = *(const uint4*)(base + (size_t)r * stride + koff);
  f.u[1] = *(const uint4*)(base + (size_t)r * stride + koff + 16);
  return f.v;
}

__global__ __launch_bounds__(256)
void f32_to_bf16_kernel(const float* __restrict__ in, unsigned short* __restrict__ out, int n) {
  int i = blockIdx.x * blockDim.x + threadIdx.x;
  if (i < n) out[i] = f2bf(in[i]);
}

// W[K,N] f32 -> Wt[N,K] bf16 (one-time, so GEMM B-staging is coalesced b128 copies)
__global__ __launch_bounds__(256)
void wt_transpose_bf16(const float* __restrict__ in, unsigned short* __restrict__ out,
                       int K, int N) {
  __shared__ float t[16][17];
  int k0 = blockIdx.y * 16, n0 = blockIdx.x * 16;
  int tx = threadIdx.x & 15, ty = threadIdx.x >> 4;
  t[ty][tx] = in[(size_t)(k0 + ty) * N + n0 + tx];
  __syncthreads();
  out[(size_t)(n0 + ty) * K + k0 + tx] = f2bf(t[tx][ty]);
}

// C = A[M,K] * Bt[N,K]^T + bias, bf16 in, f32 WMMA accumulate, async double-buffered LDS.
// mode 0: f32 -> outF[M,N]
// mode 1: QKV scatter -> Q,K as [B,H,T,D], V as [B,H,D,T]; Q scaled by 1/sqrt(D)
__global__ __launch_bounds__(256)
void gemm_bf16_wmma(const unsigned short* __restrict__ A,
                    const unsigned short* __restrict__ Bt,
                    const float* __restrict__ bias,
                    float* __restrict__ outF,
                    unsigned short* __restrict__ qkvOut,
                    int M, int N, int K, int mode)
{
  __shared__ unsigned short sA[2][BM * LPAD];
  __shared__ unsigned short sB[2][BN * LPAD];

  const int tid  = threadIdx.x;
  const int lane = tid & 31;
  const int wave = tid >> 5;            // 8 waves
  const int wm0  = (wave >> 1) * 32;    // 4 along M
  const int wn0  = (wave & 1) * 64;     // 2 along N
  const int m0 = blockIdx.y * BM;
  const int n0 = blockIdx.x * BN;

  const int sr = tid >> 2;              // staging row 0..63 (+64 second pass)
  const int sc = (tid & 3) * 8;         // 8 bf16 per b128

  v8f acc[2][4];
  #pragma unroll
  for (int i = 0; i < 2; ++i)
    #pragma unroll
    for (int j = 0; j < 4; ++j)
      acc[i][j] = (v8f){0.f,0.f,0.f,0.f,0.f,0.f,0.f,0.f};

  const int nkt = K / BK;

  // stage tile 0 into buffer 0
  #pragma unroll
  for (int p = 0; p < 2; ++p) {
    cp16_g2l(A  + (size_t)(m0 + sr + p * 64) * K + sc, &sA[0][(sr + p * 64) * LPAD + sc]);
    cp16_g2l(Bt + (size_t)(n0 + sr + p * 64) * K + sc, &sB[0][(sr + p * 64) * LPAD + sc]);
  }

  for (int kt = 0; kt < nkt; ++kt) {
    const int cur = kt & 1;
    if (kt + 1 < nkt) {
      const int kn = (kt + 1) * BK;
      #pragma unroll
      for (int p = 0; p < 2; ++p) {
        cp16_g2l(A  + (size_t)(m0 + sr + p * 64) * K + kn + sc,
                 &sA[cur ^ 1][(sr + p * 64) * LPAD + sc]);
        cp16_g2l(Bt + (size_t)(n0 + sr + p * 64) * K + kn + sc,
                 &sB[cur ^ 1][(sr + p * 64) * LPAD + sc]);
      }
      if (kt + 2 < nkt) {
        __builtin_prefetch(A  + (size_t)(m0 + sr) * K + kn + BK + sc, 0, 3);
        __builtin_prefetch(Bt + (size_t)(n0 + sr) * K + kn + BK + sc, 0, 3);
      }
      WAIT_ASYNC(4);   // next tile may stay in flight; current tile (oldest 4) landed
    } else {
      WAIT_ASYNC(0);
    }
    __syncthreads();

    v16bf af[2], bfr[4];
    #pragma unroll
    for (int mi = 0; mi < 2; ++mi) af[mi]  = frag_ld(sA[cur], wm0 + mi * 16, LPAD, lane);
    #pragma unroll
    for (int ni = 0; ni < 4; ++ni) bfr[ni] = frag_ld(sB[cur], wn0 + ni * 16, LPAD, lane);

    #pragma unroll
    for (int mi = 0; mi < 2; ++mi)
      #pragma unroll
      for (int ni = 0; ni < 4; ++ni)
        acc[mi][ni] = __builtin_amdgcn_wmma_f32_16x16x32_bf16(
            false, af[mi], false, bfr[ni], (short)0, acc[mi][ni], false, false);
    __syncthreads();   // all waves done with buf[cur] before it is restaged
  }

  // ---- epilogue ----
  const int rbase = (lane >> 4) << 3;
  const int cl    = lane & 15;
  const size_t HD = (size_t)B_ * H_ * T_ * D_;
  #pragma unroll
  for (int mi = 0; mi < 2; ++mi)
    #pragma unroll
    for (int ni = 0; ni < 4; ++ni)
      #pragma unroll
      for (int r = 0; r < 8; ++r) {
        int row = m0 + wm0 + mi * 16 + rbase + r;
        int col = n0 + wn0 + ni * 16 + cl;
        float v = acc[mi][ni][r] + bias[col];
        if (mode == 0) {
          outF[(size_t)row * N + col] = v;
        } else {
          int s = col >> 10, h = (col >> 6) & 15, d = col & 63;
          int b = row >> 11, t = row & 2047;
          size_t off;
          if (s == 0)      { v *= 0.125f; off = ((size_t)(b * H_ + h) * T_ + t) * D_ + d; }
          else if (s == 1) { off = HD +     ((size_t)(b * H_ + h) * T_ + t) * D_ + d; }
          else             { off = 2 * HD + ((size_t)(b * H_ + h) * D_ + d) * T_ + t; }
          qkvOut[off] = f2bf(v);
        }
      }
}

// Causal flash attention. Q,K: [B,H,T,D] bf16. V: [B,H,D,T] bf16. Out: bf16 [B*T, C].
// Grid (T/64, B*H), block 128. No block barriers: all cross-lane data is wave-private
// (LDS ops of one wave complete in order), so each wave runs its exact causal tile count.
__global__ __launch_bounds__(128)
void attn_wmma(const unsigned short* __restrict__ Q,
               const unsigned short* __restrict__ Kb,
               const unsigned short* __restrict__ Vt,
               unsigned short* __restrict__ attnOut)
{
  __shared__ unsigned short sP[4][16 * LPAD];   // per-wave P scratch (D->A layout)

  const int tid = threadIdx.x, lane = tid & 31, wave = tid >> 5;
  const int bh = blockIdx.y;
  const int b = bh >> 4, h = bh & 15;
  const int q0 = blockIdx.x * 64;
  const size_t baseQ = (size_t)bh * T_ * D_;   // also K base; V base same (D*T == T*D)

  const int cl   = lane & 15;
  const int koff = (lane >> 4) * 8;
  const int rhi  = (lane >> 4) << 3;

  // Q fragments (16x64 -> two 16x32 A-fragments), 1/sqrt(D) already folded in
  v16bf qf[2];
  {
    const unsigned short* qp = Q + baseQ + (size_t)(q0 + wave * 16 + cl) * D_;
    #pragma unroll
    for (int s = 0; s < 2; ++s) {
      union { uint4 u[2]; v16bf v; } f;
      f.u[0] = *(const uint4*)(qp + s * 32 + koff);
      f.u[1] = *(const uint4*)(qp + s * 32 + koff + 16);
      qf[s] = f.v;
    }
  }

  v8f oacc[4];
  #pragma unroll
  for (int i = 0; i < 4; ++i) oacc[i] = (v8f){0.f,0.f,0.f,0.f,0.f,0.f,0.f,0.f};
  float mrow[8], lrow[8];
  #pragma unroll
  for (int r = 0; r < 8; ++r) { mrow[r] = -3.0e38f; lrow[r] = 0.f; }

  // exact causal coverage for this wave's 16 rows
  const int nkt = ((q0 + wave * 16 + 15) >> 5) + 1;
  for (int kt = 0; kt < nkt; ++kt) {
    const int kb0 = kt * 32;

    if (kt + 1 < nkt) {
      __builtin_prefetch(Kb + baseQ + (size_t)(kb0 + 32 + cl) * D_ + koff, 0, 3);
      __builtin_prefetch(Vt + baseQ + (size_t)cl * T_ + kb0 + 32 + koff, 0, 3);
    }

    // ---- S = Q K^T : K fragments direct from global (d-contiguous rows) ----
    v8f sacc[2];
    sacc[0] = (v8f){0.f,0.f,0.f,0.f,0.f,0.f,0.f,0.f};
    sacc[1] = (v8f){0.f,0.f,0.f,0.f,0.f,0.f,0.f,0.f};
    #pragma unroll
    for (int nb = 0; nb < 2; ++nb) {
      const unsigned short* kp = Kb + baseQ + (size_t)(kb0 + nb * 16 + cl) * D_;
      #pragma unroll
      for (int s = 0; s < 2; ++s) {
        union { uint4 u[2]; v16bf v; } f;
        f.u[0] = *(const uint4*)(kp + s * 32 + koff);
        f.u[1] = *(const uint4*)(kp + s * 32 + koff + 16);
        sacc[nb] = __builtin_amdgcn_wmma_f32_16x16x32_bf16(
            false, qf[s], false, f.v, (short)0, sacc[nb], false, false);
      }
    }

    // ---- causal mask + online softmax ----
    #pragma unroll
    for (int r = 0; r < 8; ++r) {
      int q = q0 + wave * 16 + rhi + r;
      float s0 = sacc[0][r], s1 = sacc[1][r];
      if (kb0 + cl      > q) s0 = -3.0e38f;
      if (kb0 + 16 + cl > q) s1 = -3.0e38f;
      float lm = fmaxf(s0, s1);
      #pragma unroll
      for (int off = 8; off >= 1; off >>= 1) lm = fmaxf(lm, __shfl_xor(lm, off, 32));
      float mn = fmaxf(mrow[r], lm);
      float p0 = __expf(s0 - mn);
      float p1 = __expf(s1 - mn);
      float alpha = __expf(mrow[r] - mn);
      float ls = p0 + p1;
      #pragma unroll
      for (int off = 8; off >= 1; off >>= 1) ls += __shfl_xor(ls, off, 32);
      lrow[r] = lrow[r] * alpha + ls;
      mrow[r] = mn;
      #pragma unroll
      for (int nd = 0; nd < 4; ++nd) oacc[nd][r] = oacc[nd][r] * alpha;
      sP[wave][(rhi + r) * LPAD + cl]      = f2bf(p0);
      sP[wave][(rhi + r) * LPAD + 16 + cl] = f2bf(p1);
    }
    asm volatile("" ::: "memory");   // keep LDS store->load order in the scheduler

    // ---- O += P V : V fragments direct from global [B,H,D,T] (key-contiguous) ----
    v16bf pf = frag_ld(sP[wave], 0, LPAD, lane);
    #pragma unroll
    for (int nd = 0; nd < 4; ++nd) {
      const unsigned short* vp = Vt + baseQ + (size_t)(nd * 16 + cl) * T_ + kb0;
      union { uint4 u[2]; v16bf v; } f;
      f.u[0] = *(const uint4*)(vp + koff);
      f.u[1] = *(const uint4*)(vp + koff + 16);
      oacc[nd] = __builtin_amdgcn_wmma_f32_16x16x32_bf16(
          false, pf, false, f.v, (short)0, oacc[nd], false, false);
    }
  }

  // ---- normalize + store to [B*T, C] bf16 ----
  #pragma unroll
  for (int nd = 0; nd < 4; ++nd)
    #pragma unroll
    for (int r = 0; r < 8; ++r) {
      int t = q0 + wave * 16 + rhi + r;
      int d = nd * 16 + cl;
      float v = oacc[nd][r] / lrow[r];
      attnOut[(size_t)(b * T_ + t) * C_ + h * D_ + d] = f2bf(v);
    }
}

extern "C" void kernel_launch(void* const* d_in, const int* in_sizes, int n_in,
                              void* d_out, int out_size, void* d_ws, size_t ws_size,
                              hipStream_t stream)
{
  const float* x    = (const float*)d_in[0];
  // d_in[1] = causal mask (bool): recomputed analytically in-kernel
  const float* Wqkv = (const float*)d_in[2];
  const float* bqkv = (const float*)d_in[3];
  const float* Wout = (const float*)d_in[4];
  const float* bout = (const float*)d_in[5];
  float* out = (float*)d_out;

  const size_t MT = (size_t)B_ * T_;   // 8192 rows
  char* ws = (char*)d_ws;
  unsigned short* xb    = (unsigned short*)ws; ws += MT * C_ * 2;
  unsigned short* wqkvb = (unsigned short*)ws; ws += (size_t)C_ * 3 * C_ * 2;  // [3C][C] transposed
  unsigned short* woutb = (unsigned short*)ws; ws += (size_t)C_ * C_ * 2;      // [C][C] transposed
  unsigned short* qkvb  = (unsigned short*)ws; ws += 3 * MT * C_ * 2;          // Q,K [B,H,T,D]; V [B,H,D,T]
  unsigned short* attnb = (unsigned short*)ws; ws += MT * C_ * 2;

  int n1 = (int)(MT * C_);
  f32_to_bf16_kernel<<<(n1 + 255) / 256, 256, 0, stream>>>(x, xb, n1);
  wt_transpose_bf16<<<dim3(3 * C_ / 16, C_ / 16), 256, 0, stream>>>(Wqkv, wqkvb, C_, 3 * C_);
  wt_transpose_bf16<<<dim3(C_ / 16, C_ / 16), 256, 0, stream>>>(Wout, woutb, C_, C_);

  // QKV projection: [8192,1024] x [1024,3072] -> Q/K/V bf16 (V transposed)
  dim3 g1(3 * C_ / BN, (unsigned)(MT / BM));
  gemm_bf16_wmma<<<g1, 256, 0, stream>>>(xb, wqkvb, bqkv, nullptr, qkvb,
                                         (int)MT, 3 * C_, C_, 1);

  // causal flash attention
  dim3 g2(T_ / 64, B_ * H_);
  attn_wmma<<<g2, 128, 0, stream>>>(qkvb, qkvb + MT * C_, qkvb + 2 * MT * C_, attnb);

  // output projection: [8192,1024] x [1024,1024] + bout -> f32 d_out
  dim3 g3(C_ / BN, (unsigned)(MT / BM));
  gemm_bf16_wmma<<<g3, 256, 0, stream>>>(attnb, woutb, bout, out, nullptr,
                                         (int)MT, C_, C_, 0);
}
